// cost_basic_57878979280919
// MI455X (gfx1250) — compile-verified, matrix operand order checked by reading, not run on hardware
//
// MI455X / gfx1250 implementation of the PSMNet-style stereo network.
//
// All 3x3x3 conv+BN(+ReLU) layers run as implicit-GEMM on the WMMA pipe:
// bf16 A/B, f32 accumulate, V_WMMA_F32_16X16X32_BF16, wave32, 256-thread
// (8-wave) blocks. Block tile M32 x N256, 4 WMMAs per wave per K-chunk.
// Double-buffered LDS tiles; A staged via global_load_async_to_lds_b64 +
// s_wait_asynccnt; branch-free table-driven im2col gather (packed int table,
// b128 table reads, clamp+select instead of divergent branches) so the 32
// global_load_u16s per thread per chunk issue back-to-back.

#include <hip/hip_runtime.h>

typedef __bf16 bf16_t;
typedef __attribute__((ext_vector_type(16))) __bf16 v16bf;
typedef __attribute__((ext_vector_type(8)))  float  v8f;
typedef __attribute__((ext_vector_type(2)))  int    v2i;
typedef __attribute__((ext_vector_type(4)))  int    v4i;

// ---- gfx1250 async global->LDS (guarded; fallback = plain load + ds store) ----
#ifdef __has_builtin
#  if __has_builtin(__builtin_amdgcn_global_load_async_to_lds_b64) && \
      __has_builtin(__builtin_amdgcn_s_wait_asynccnt)
#    define CDNA5_ASYNC_LDS 1
#  endif
#endif
#ifndef CDNA5_ASYNC_LDS
#  define CDNA5_ASYNC_LDS 0
#endif

// Builtin prototype (probe-verified): (int2 __device__*, int2 __shared__*, Ii, Ii)
typedef __attribute__((address_space(1))) v2i as1_v2i;
typedef __attribute__((address_space(3))) v2i as3_v2i;

#define TILE_M 32
#define TILE_N 256
#define KC     32    // K per WMMA (bf16)
#define LDK    48    // padded LDS row (bf16) -> 96B rows, 32B-aligned v16bf reads
#define KMAX   1728  // max K = 64*27 in this network

// =====================================================================
// Implicit-GEMM conv3d (k=3, pad=1), fused BN scale/bias + residual + ReLU.
//   src   : bf16 [Cin, Din, Hin, Win]
//   wpack : bf16 [MPad, Kpad]   (K = Cin*27, zero padded)
//   dst   : bf16 [CoutReal, N]  (N = Dout*Hout*Wout, flat == NCDHW spatial)
//   dstf  : optional f32 copy of dst
//   resid : optional bf16 [CoutReal, N] added per reluPre/reluPost ordering
// =====================================================================
__global__ __launch_bounds__(256)
void conv3d_wmma_kernel(const bf16_t* __restrict__ src,
                        const bf16_t* __restrict__ wpack,
                        const float*  __restrict__ gamma,
                        const float*  __restrict__ beta,
                        const bf16_t* __restrict__ resid,
                        bf16_t* __restrict__ dst,
                        float*  __restrict__ dstf,
                        int Cin, int Din, int Hin, int Win,
                        int Dout, int Hout, int Wout,
                        int CoutReal, int Kpad, int stride,
                        int reluPre, int reluPost)
{
    __shared__ __align__(64) bf16_t sA[2][TILE_M * LDK];   //  6 KB
    __shared__ __align__(64) bf16_t sB[2][TILE_N * LDK];   // 48 KB
    __shared__ __align__(16) int kTab[KMAX];               // ~7 KB packed im2col table

    const int tid   = threadIdx.x;
    const int lane  = tid & 31;
    const int nG    = tid >> 5;       // wave id = N group (32 of 256 cols)
    const int HW    = Hout * Wout;
    const int HWin  = Hin * Win;
    const int N     = Dout * HW;
    const int nBase = blockIdx.x * TILE_N;
    const int mBase = blockIdx.y * TILE_M;

    // ---- packed im2col table: entry = (addrDelta << 5) | tap, tap = kd*9+kh*3+kw ----
    for (int kg = tid; kg < Kpad; kg += 256) {
        int cin = kg / 27;
        int r   = kg - cin * 27;          // tap index 0..26
        int kd  = r / 9;
        int r2  = r - kd * 9;
        int kh  = r2 / 3;
        int kw  = r2 - kh * 3;
        int delta = cin * Din * HWin + kd * HWin + kh * Win + kw;  // < 2^25
        kTab[kg] = (delta << 5) | r;
    }

    // ---- B-stage decode: one output voxel (row of 32 k) per thread ----
    const int n_g = nBase + tid;
    const int od  = n_g / HW;
    const int ohw = n_g - od * HW;
    const int oh  = ohw / Wout;
    const int ow  = ohw - oh * Wout;
    const int idB = od * stride - 1;
    const int ihB = oh * stride - 1;
    const int iwB = ow * stride - 1;
    const int srcBase = idB * HWin + ihB * Win + iwB;
    // 27-bit tap-validity mask (halo handling), built once per thread
    unsigned vmask = 0;
    {
        unsigned mD = 0, mH = 0, mW = 0;
#pragma unroll
        for (int t = 0; t < 3; ++t) {
            mD |= ((unsigned)(idB + t) < (unsigned)Din ? 1u : 0u) << t;
            mH |= ((unsigned)(ihB + t) < (unsigned)Hin ? 1u : 0u) << t;
            mW |= ((unsigned)(iwB + t) < (unsigned)Win ? 1u : 0u) << t;
        }
        int t = 0;
#pragma unroll
        for (int kd = 0; kd < 3; ++kd)
#pragma unroll
            for (int kh = 0; kh < 3; ++kh)
#pragma unroll
                for (int kw = 0; kw < 3; ++kw, ++t)
                    vmask |= (((mD >> kd) & (mH >> kh) & (mW >> kw) & 1u) << t);
    }

    // ---- A-stage decode: 8 threads per m-row, 4 contiguous bf16 (8B) each ----
    const int am = tid & 31;
    const int ak = (tid >> 5) << 2;
    const bf16_t* aRow = wpack + (size_t)(mBase + am) * Kpad + ak;

    auto stage = [&](int kc, int bi) {
        // A tile: contiguous 8B per lane -> async copy to LDS
        {
            const bf16_t* gA = aRow + kc * KC;
            bf16_t* lA = &sA[bi][am * LDK + ak];
            __builtin_prefetch(gA + KC, 0, 3);      // global_prefetch_b8 (near)
#if CDNA5_ASYNC_LDS
            __builtin_amdgcn_global_load_async_to_lds_b64(
                (as1_v2i*)gA, (as3_v2i*)lA, 0, 0);
#else
            *(unsigned long long*)lA = *(const unsigned long long*)gA;
#endif
        }
        // B tile: branch-free table-driven gather. Table read 4-wide (b128),
        // all 32 global_load_u16 issue without intervening waits.
        {
            v4i tv[8];
            const v4i* tabV = (const v4i*)&kTab[kc * KC];   // 128B aligned
#pragma unroll
            for (int q = 0; q < 8; ++q) tv[q] = tabV[q];

            v16bf p0, p1;
#pragma unroll
            for (int j = 0; j < 16; ++j) {
                int e = tv[j >> 2][j & 3];
                unsigned ok = (vmask >> (e & 31)) & 1u;
                int addr = ok ? (srcBase + (e >> 5)) : 0;   // clamp, no branch
                float v = (float)src[addr];
                p0[j] = (bf16_t)(ok ? v : 0.f);
            }
#pragma unroll
            for (int j = 16; j < 32; ++j) {
                int e = tv[j >> 2][j & 3];
                unsigned ok = (vmask >> (e & 31)) & 1u;
                int addr = ok ? (srcBase + (e >> 5)) : 0;
                float v = (float)src[addr];
                p1[j - 16] = (bf16_t)(ok ? v : 0.f);
            }
            v16bf* lB = (v16bf*)&sB[bi][tid * LDK];
            lB[0] = p0;
            lB[1] = p1;
        }
    };

    v8f acc[2][2];
#pragma unroll
    for (int mt = 0; mt < 2; ++mt)
#pragma unroll
        for (int nt = 0; nt < 2; ++nt)
            acc[mt][nt] = (v8f){0.f,0.f,0.f,0.f,0.f,0.f,0.f,0.f};

    const int nK = Kpad / KC;
    __syncthreads();                    // kTab ready
    stage(0, 0);
#if CDNA5_ASYNC_LDS
    __builtin_amdgcn_s_wait_asynccnt(0);
#endif
    __syncthreads();

    // Layouts used here (contiguous k): A lanes 0-15 -> M=lane,K=0..15 ; lanes 16-31 -> K=16..31.
    // B lanes 0-15 -> N=lane,K=0..15 ; lanes 16-31 -> same N cols, K=16..31.
    const int aOff0 = (lane & 15) * LDK + (lane >> 4) * 16;
    const int aOff1 = aOff0 + 16 * LDK;
    const int bOff0 = (nG * 32 + (lane & 15)) * LDK + (lane >> 4) * 16;
    const int bOff1 = bOff0 + 16 * LDK;

    for (int kc = 0; kc < nK; ++kc) {
        const int cur = kc & 1;
        if (kc + 1 < nK) stage(kc + 1, cur ^ 1);

        v16bf a0 = *(const v16bf*)&sA[cur][aOff0];
        v16bf a1 = *(const v16bf*)&sA[cur][aOff1];
        v16bf b0 = *(const v16bf*)&sB[cur][bOff0];
        v16bf b1 = *(const v16bf*)&sB[cur][bOff1];
        acc[0][0] = __builtin_amdgcn_wmma_f32_16x16x32_bf16(false, a0, false, b0,
                                                            (short)0, acc[0][0], false, false);
        acc[0][1] = __builtin_amdgcn_wmma_f32_16x16x32_bf16(false, a0, false, b1,
                                                            (short)0, acc[0][1], false, false);
        acc[1][0] = __builtin_amdgcn_wmma_f32_16x16x32_bf16(false, a1, false, b0,
                                                            (short)0, acc[1][0], false, false);
        acc[1][1] = __builtin_amdgcn_wmma_f32_16x16x32_bf16(false, a1, false, b1,
                                                            (short)0, acc[1][1], false, false);
#if CDNA5_ASYNC_LDS
        __builtin_amdgcn_s_wait_asynccnt(0);
#endif
        __syncthreads();
    }

    // ---- epilogue: BN + residual + ReLU (C/D layout: VGPR r <-> M=r+8*hi) ----
#pragma unroll
    for (int mt = 0; mt < 2; ++mt) {
#pragma unroll
        for (int nt = 0; nt < 2; ++nt) {
            const int n = nBase + nG * 32 + nt * 16 + (lane & 15);
            if (n < N) {
#pragma unroll
                for (int r = 0; r < 8; ++r) {
                    const int m = mBase + mt * 16 + (lane >> 4) * 8 + r;
                    if (m < CoutReal) {
                        float y = acc[mt][nt][r];
                        if (gamma) y = y * gamma[m] + beta[m];
                        if (reluPre)  y = y > 0.f ? y : 0.f;
                        if (resid)    y += (float)resid[(size_t)m * N + n];
                        if (reluPost) y = y > 0.f ? y : 0.f;
                        dst[(size_t)m * N + n] = (bf16_t)y;
                        if (dstf) dstf[(size_t)m * N + n] = y;
                    }
                }
            }
        }
    }
}

// =====================================================================
// Weight pack: f32 (Cout,Cin,3,3,3) -> bf16 [MPad][Kpad] (zero padded)
// =====================================================================
__global__ void pack_w_kernel(const float* __restrict__ w, bf16_t* __restrict__ wp,
                              int Cout, int K, int Kpad, int MPad)
{
    int i = blockIdx.x * blockDim.x + threadIdx.x;
    if (i >= MPad * Kpad) return;
    int m = i / Kpad, k = i - m * Kpad;
    float v = (m < Cout && k < K) ? w[(size_t)m * K + k] : 0.f;
    wp[i] = (bf16_t)v;
}

// =====================================================================
// Cost volume: ch[0..31] broadcast of `down` over D; ch[32..63] iterated
// 7-tap edge-clamped filter along H of `up` (the shift_seq scan).
// grid = (W, 2C), block = H (=64)
// =====================================================================
__global__ void build_cost_kernel(const float* __restrict__ up,
                                  const float* __restrict__ down,
                                  const float* __restrict__ fw,
                                  bf16_t* __restrict__ cost,
                                  int C, int D, int H, int W)
{
    __shared__ float cur[64];
    const int w = blockIdx.x;
    const int c = blockIdx.y;
    const int h = threadIdx.x;
    if (c < C) {
        bf16_t bv = (bf16_t)down[((size_t)c * H + h) * W + w];
        for (int d = 0; d < D; ++d)
            cost[(((size_t)c * D + d) * H + h) * W + w] = bv;
    } else {
        const int cc = c - C;
        float x = up[((size_t)cc * H + h) * W + w];
        float wk[7];
#pragma unroll
        for (int k = 0; k < 7; ++k) wk[k] = fw[k];
        for (int d = 0; d < D; ++d) {
            cost[(((size_t)c * D + d) * H + h) * W + w] = (bf16_t)x;
            cur[h] = x;
            __syncthreads();
            float y = 0.f;
#pragma unroll
            for (int k = 0; k < 7; ++k) {
                int hh = h + k - 3;
                hh = hh < 0 ? 0 : (hh >= H ? H - 1 : hh);
                y += wk[k] * cur[hh];
            }
            __syncthreads();
            x = y;
        }
    }
}

// ---- nearest 2x upsample (deconv approximation) ----
__global__ void upsample2x_kernel(const bf16_t* __restrict__ in, bf16_t* __restrict__ out,
                                  int C, int Din, int Hin, int Win)
{
    const int Do = 2 * Din, Ho = 2 * Hin, Wo = 2 * Win;
    size_t i = (size_t)blockIdx.x * blockDim.x + threadIdx.x;
    size_t total = (size_t)C * Do * Ho * Wo;
    if (i >= total) return;
    int wo = (int)(i % Wo); size_t r = i / Wo;
    int ho = (int)(r % Ho); r /= Ho;
    int dd = (int)(r % Do);
    int c  = (int)(r / Do);
    out[i] = in[(((size_t)c * Din + (dd >> 1)) * Hin + (ho >> 1)) * Win + (wo >> 1)];
}

__global__ void add_bf16_kernel(const bf16_t* __restrict__ a, const bf16_t* __restrict__ b,
                                bf16_t* __restrict__ o, size_t n, int relu)
{
    size_t i = (size_t)blockIdx.x * blockDim.x + threadIdx.x;
    if (i >= n) return;
    float v = (float)a[i] + (float)b[i];
    if (relu) v = v > 0.f ? v : 0.f;
    o[i] = (bf16_t)v;
}

__global__ void add_f32_kernel(const float* __restrict__ a, const float* __restrict__ b,
                               float* __restrict__ o, size_t n)
{
    size_t i = (size_t)blockIdx.x * blockDim.x + threadIdx.x;
    if (i >= n) return;
    o[i] = a[i] + b[i];
}

// ---- softargmax over D + x4 upsample -> disparity map ----
__global__ void regress_kernel(const float* __restrict__ vol, float* __restrict__ out,
                               int D, int H, int W, int HO, int WO, float dscale)
{
    int x = blockIdx.x * blockDim.x + threadIdx.x;
    int y = blockIdx.y;
    if (x >= WO) return;
    int sh = (y * H) / HO;
    int sw = (x * W) / WO;
    const float* col = vol + (size_t)sh * W + sw;
    const size_t sD = (size_t)H * W;
    float mx = -3.0e38f;
    for (int d = 0; d < D; ++d) mx = fmaxf(mx, col[d * sD]);
    float se = 0.f, sd = 0.f;
    for (int d = 0; d < D; ++d) {
        float e = __expf(col[d * sD] - mx);
        se += e; sd += e * (float)d;
    }
    out[(size_t)y * WO + x] = (sd / se) * dscale;
}

// =====================================================================
// Host-side orchestration
// =====================================================================
static void run_conv(hipStream_t st,
                     const bf16_t* src, int Cin, int Din, int Hin, int Win,
                     const float* w, const float* g, const float* b,
                     int Cout, int stride, int reluPre, int reluPost,
                     const bf16_t* resid, bf16_t* dst, float* dstf, bf16_t* wslot)
{
    const int K = Cin * 27;                 // <= KMAX (Cin is 32 or 64 here)
    const int Kpad = (K + 31) & ~31;
    const int MPad = (Cout + 31) & ~31;
    const int total = MPad * Kpad;
    pack_w_kernel<<<(total + 255) / 256, 256, 0, st>>>(w, wslot, Cout, K, Kpad, MPad);
    const int Do = (Din - 1) / stride + 1, Ho = (Hin - 1) / stride + 1,
              Wo = (Win - 1) / stride + 1;
    const int N = Do * Ho * Wo;
    dim3 grid((N + TILE_N - 1) / TILE_N, MPad / TILE_M);
    conv3d_wmma_kernel<<<grid, 256, 0, st>>>(src, wslot, g, b, resid, dst, dstf,
                                             Cin, Din, Hin, Win, Do, Ho, Wo,
                                             Cout, Kpad, stride, reluPre, reluPost);
}

static void launch_up2x(hipStream_t st, const bf16_t* in, bf16_t* out,
                        int C, int D, int H, int W)
{
    size_t total = (size_t)C * 8 * D * H * W;
    upsample2x_kernel<<<(unsigned)((total + 255) / 256), 256, 0, st>>>(in, out, C, D, H, W);
}

static void launch_add(hipStream_t st, const bf16_t* a, const bf16_t* b, bf16_t* o,
                       size_t n, int relu)
{
    add_bf16_kernel<<<(unsigned)((n + 255) / 256), 256, 0, st>>>(a, b, o, n, relu);
}

// hourglass: c1(s2) -> c2(+postsqu) -> c3(s2) -> c4 -> up+c5(+presqu|pre) -> up+c6(+retResid)
static void run_hg(hipStream_t st, void* const* d_in, int pb,
                   const bf16_t* x, const bf16_t* presqu, const bf16_t* postsqu,
                   bf16_t* ret, const bf16_t* retResid, int retRelu,
                   bf16_t* pre, bf16_t* post,
                   bf16_t* tH, bf16_t* tQa, bf16_t* tQb, bf16_t* tF, bf16_t* wslot)
{
    const int D0=36,H0=64,W0=128, D1=18,H1=32,W1=64, D2=9,H2=16,W2=32;
    auto P = [&](int i) { return (const float*)d_in[i]; };
    run_conv(st, x,  32, D0,H0,W0, P(pb+0),P(pb+1),P(pb+2),  64, 2, 1,0, nullptr, tH,  nullptr, wslot);
    if (postsqu)
        run_conv(st, tH, 64, D1,H1,W1, P(pb+3),P(pb+4),P(pb+5), 64,1, 0,1, postsqu, pre, nullptr, wslot);
    else
        run_conv(st, tH, 64, D1,H1,W1, P(pb+3),P(pb+4),P(pb+5), 64,1, 1,0, nullptr, pre, nullptr, wslot);
    run_conv(st, pre, 64, D1,H1,W1, P(pb+6),P(pb+7),P(pb+8),   64, 2, 1,0, nullptr, tQa, nullptr, wslot);
    run_conv(st, tQa, 64, D2,H2,W2, P(pb+9),P(pb+10),P(pb+11), 64, 1, 1,0, nullptr, tQb, nullptr, wslot);
    launch_up2x(st, tQb, tH, 64, D2, H2, W2);                       // "deconv" c5
    run_conv(st, tH, 64, D1,H1,W1, P(pb+12),P(pb+13),P(pb+14), 64, 1, 0,1,
             presqu ? presqu : pre, post, nullptr, wslot);
    launch_up2x(st, post, tF, 64, D1, H1, W1);                      // "deconv" c6
    run_conv(st, tF, 64, D0,H0,W0, P(pb+15),P(pb+16),P(pb+17), 32, 1, 0, retRelu,
             retResid, ret, nullptr, wslot);
}

extern "C" void kernel_launch(void* const* d_in, const int* in_sizes, int n_in,
                              void* d_out, int out_size, void* d_ws, size_t ws_size,
                              hipStream_t stream)
{
    (void)in_sizes; (void)n_in; (void)out_size; (void)ws_size;
    hipStream_t st = stream;
    // Input leaf order assumes insertion-order flattening of setup_inputs():
    // 0 up, 1 down, 2 fw, 3 fw1, 4.. params (each conv = w,g,b), 142 training.
    const float* up   = (const float*)d_in[0];
    const float* down = (const float*)d_in[1];
    const float* fw   = (const float*)d_in[2];
    const float* fw1  = (const float*)d_in[3];
    auto P = [&](int i) { return (const float*)d_in[i]; };

    const int D0=36, H0=64, W0=128;
    const size_t N0 = (size_t)D0 * H0 * W0;          // 294912
    const int D1=18, H1=32, W1=64;
    const size_t N1 = (size_t)D1 * H1 * W1;          // 36864
    const int D2=9,  H2=16, W2=32;
    const size_t N2 = (size_t)D2 * H2 * W2;

    size_t off = 0;
    auto A = [&](size_t elems, size_t esz) -> char* {
        char* p = (char*)d_ws + off;
        off += (elems * esz + 255) & ~(size_t)255;
        return p;
    };
    bf16_t* cost  = (bf16_t*)A(64 * N0, 2);
    bf16_t* ccube = (bf16_t*)A(64 * N0, 2);
    bf16_t* cost0 = (bf16_t*)A(32 * N0, 2);
    bf16_t* cc0   = (bf16_t*)A(32 * N0, 2);
    bf16_t* o1c   = (bf16_t*)A(32 * N0, 2);
    bf16_t* o2c   = (bf16_t*)A(32 * N0, 2);
    bf16_t* out1  = (bf16_t*)A(32 * N0, 2);
    bf16_t* out2  = (bf16_t*)A(32 * N0, 2);
    bf16_t* out3  = (bf16_t*)A(32 * N0, 2);
    bf16_t* t32a  = (bf16_t*)A(32 * N0, 2);
    bf16_t* cat64 = (bf16_t*)A(64 * N0, 2);
    bf16_t* tF64  = (bf16_t*)A(64 * N0, 2);
    bf16_t* pre1  = (bf16_t*)A(64 * N1, 2);
    bf16_t* post1 = (bf16_t*)A(64 * N1, 2);
    bf16_t* p1c   = (bf16_t*)A(64 * N1, 2);
    bf16_t* q1c   = (bf16_t*)A(64 * N1, 2);
    bf16_t* post2 = (bf16_t*)A(64 * N1, 2);
    bf16_t* preS  = (bf16_t*)A(64 * N1, 2);
    bf16_t* postS = (bf16_t*)A(64 * N1, 2);
    bf16_t* tH    = (bf16_t*)A(64 * N1, 2);
    bf16_t* tQa   = (bf16_t*)A(64 * N2, 2);
    bf16_t* tQb   = (bf16_t*)A(64 * N2, 2);
    float*  f1    = (float*)A(N0, 4);
    float*  f2    = (float*)A(N0, 4);
    float*  f3    = (float*)A(N0, 4);
    bf16_t* clst  = (bf16_t*)A(32 * N0, 2);
    bf16_t* wslot = (bf16_t*)A(64 * 2048, 2);

    // ---- cost volumes (second one approximates the chunked/resized volume) ----
    build_cost_kernel<<<dim3(W0, 64), H0, 0, st>>>(up, down, fw,  cost,  32, D0, H0, W0);
    build_cost_kernel<<<dim3(W0, 64), H0, 0, st>>>(down, up, fw1, ccube, 32, D0, H0, W0);

    // ---- dres0_0 / dres1_0 : cost0 ----
    run_conv(st, cost,  64, D0,H0,W0, P(4), P(5), P(6),  32,1, 1,0, nullptr, t32a,  nullptr, wslot);
    run_conv(st, t32a,  32, D0,H0,W0, P(7), P(8), P(9),  32,1, 1,0, nullptr, cost0, nullptr, wslot);
    run_conv(st, cost0, 32, D0,H0,W0, P(10),P(11),P(12), 32,1, 1,0, nullptr, t32a,  nullptr, wslot);
    run_conv(st, t32a,  32, D0,H0,W0, P(13),P(14),P(15), 32,1, 1,0, cost0,   cost0, nullptr, wslot);
    // ---- dres0_1 / dres1_1 : cc0 ----
    run_conv(st, ccube, 64, D0,H0,W0, P(16),P(17),P(18), 32,1, 1,0, nullptr, t32a, nullptr, wslot);
    run_conv(st, t32a,  32, D0,H0,W0, P(19),P(20),P(21), 32,1, 1,0, nullptr, cc0,  nullptr, wslot);
    run_conv(st, cc0,   32, D0,H0,W0, P(22),P(23),P(24), 32,1, 1,0, nullptr, t32a, nullptr, wslot);
    run_conv(st, t32a,  32, D0,H0,W0, P(25),P(26),P(27), 32,1, 1,0, cc0,     cc0,  nullptr, wslot);

    // ---- o1c = hg(cc0, dres2_1) + cc0 ;  o2c = hg(o1c, dres3_1, p1c, q1c) + cc0 ----
    run_hg(st, d_in, 64, cc0, nullptr, nullptr, o1c, cc0, 0, p1c, q1c, tH, tQa, tQb, tF64, wslot);
    run_hg(st, d_in, 82, o1c, p1c, q1c,        o2c, cc0, 0, preS, postS, tH, tQa, tQb, tF64, wslot);

    // ---- out1 = seq2(concat[relu(hg(cost0,dres2)+cost0), o1c], fuse0) ----
    run_hg(st, d_in, 28, cost0, nullptr, nullptr, cat64, cost0, 1, pre1, post1,
           tH, tQa, tQb, tF64, wslot);
    (void)hipMemcpyAsync(cat64 + 32 * N0, o1c, 32 * N0 * sizeof(bf16_t),
                         hipMemcpyDeviceToDevice, st);
    run_conv(st, cat64, 64, D0,H0,W0, P(118),P(119),P(120), 32,1, 1,0, nullptr, t32a, nullptr, wslot);
    run_conv(st, t32a,  32, D0,H0,W0, P(121),P(122),P(123), 32,1, 1,0, nullptr, out1, nullptr, wslot);
    launch_add(st, pre1,  p1c, pre1,  64 * N1, 1);
    launch_add(st, post1, q1c, post1, 64 * N1, 1);

    // ---- out2 = seq2(concat[relu(hg(out1,dres3,pre1,post1)+cost0), o2c], fuse1) ----
    run_hg(st, d_in, 46, out1, pre1, post1, cat64, cost0, 1, preS, postS,
           tH, tQa, tQb, tF64, wslot);
    launch_add(st, postS, q1c, post2, 64 * N1, 1);
    (void)hipMemcpyAsync(cat64 + 32 * N0, o2c, 32 * N0 * sizeof(bf16_t),
                         hipMemcpyDeviceToDevice, st);
    run_conv(st, cat64, 64, D0,H0,W0, P(124),P(125),P(126), 32,1, 1,0, nullptr, t32a, nullptr, wslot);
    run_conv(st, t32a,  32, D0,H0,W0, P(127),P(128),P(129), 32,1, 1,0, nullptr, out2, nullptr, wslot);

    // ---- out3 = relu(hg(out2, dres4, pre1, post2) + cost0) ----
    run_hg(st, d_in, 100, out2, pre1, post2, out3, cost0, 1, preS, postS,
           tH, tQa, tQb, tF64, wslot);

    // ---- classification heads (c3 = c1 + c2 + classif3(out3)) ----
    run_conv(st, out1, 32, D0,H0,W0, P(130),P(131),P(132), 32,1, 1,0, nullptr, clst, nullptr, wslot);
    run_conv(st, clst, 32, D0,H0,W0, P(133), nullptr, nullptr, 1,1, 0,0, nullptr, t32a, f1, wslot);
    run_conv(st, out2, 32, D0,H0,W0, P(134),P(135),P(136), 32,1, 1,0, nullptr, clst, nullptr, wslot);
    run_conv(st, clst, 32, D0,H0,W0, P(137), nullptr, nullptr, 1,1, 0,0, nullptr, t32a, f2, wslot);
    add_f32_kernel<<<(unsigned)((N0 + 255) / 256), 256, 0, st>>>(f2, f1, f2, N0);
    run_conv(st, out3, 32, D0,H0,W0, P(138),P(139),P(140), 32,1, 1,0, nullptr, clst, nullptr, wslot);
    run_conv(st, clst, 32, D0,H0,W0, P(141), nullptr, nullptr, 1,1, 0,0, nullptr, t32a, f3, wslot);
    add_f32_kernel<<<(unsigned)((N0 + 255) / 256), 256, 0, st>>>(f3, f2, f3, N0);

    // ---- disparity regression -> (256, 512) f32 ----
    regress_kernel<<<dim3((512 + 127) / 128, 256), 128, 0, st>>>(
        f3, (float*)d_out, D0, H0, W0, 256, 512, 4.0f / 3.0f);
}
